// GATNet_67181878444859
// MI455X (gfx1250) — matrix operand order repaired and checked by reading, not compiled, not run on hardware
//
#include <hip/hip_runtime.h>
#include <hip/hip_bf16.h>

// ---------------- problem constants ----------------
#define N_NODES  50000
#define N_EDGES  800000
#define E_TOT    850000          // edges + self loops
#define IN_CH    128
#define HID      32
#define HEADS    8
#define HIDALL   256             // HEADS*HID
#define OUT_CH   16
#define NEG_SLOPE 0.2f
#define MTILES   (N_NODES/16)    // 3125 (N_NODES divisible by 16)

typedef __bf16 bf16_t;
typedef __attribute__((ext_vector_type(16))) __bf16 v16bf;
typedef __attribute__((ext_vector_type(8)))  float  v8f;

__device__ __forceinline__ v8f wmma_bf16(v16bf a, v16bf b, v8f c) {
  // D = A(16x32 bf16) x B(32x16 bf16) + C(16x16 f32)
  return __builtin_amdgcn_wmma_f32_16x16x32_bf16(false, a, false, b, (short)0, c,
                                                 false, false);
}

// monotonic float <-> orderable-uint mapping for atomicMax-based segment max
__device__ __forceinline__ unsigned f2ord(float f) {
  unsigned u = __float_as_uint(f);
  return (u & 0x80000000u) ? ~u : (u | 0x80000000u);
}
__device__ __forceinline__ float ord2f(unsigned u) {
  unsigned v = (u & 0x80000000u) ? (u & 0x7FFFFFFFu) : ~u;
  return __uint_as_float(v);
}

__device__ __forceinline__ void edge_nodes(const int* __restrict__ ei, int e,
                                           int& s, int& d) {
  if (e < N_EDGES) { s = ei[e]; d = ei[N_EDGES + e]; }
  else             { s = d = e - N_EDGES; }   // self loop
}

// ---------------- operand pre-swizzle (ISA 7.12.2 layouts) ----------------
// A-fragment (16x32 bf16): lane l(0..15)/hi, elem i -> k = ((i<8)?i:i+8)+hi*8
__global__ void k_xswz(const float* __restrict__ x, bf16_t* __restrict__ xs) {
  int t = blockIdx.x * blockDim.x + threadIdx.x;
  const int total = MTILES * (IN_CH / 32) * 32;            // 400000
  if (t >= total) return;
  int lane = t & 31, kc = (t >> 5) & 3, mt = t >> 7;
  int l = lane & 15, hi = lane >> 4;
  const float* xr = x + (size_t)(mt * 16 + l) * IN_CH + kc * 32;
  bf16_t* o = xs + ((size_t)(mt * 4 + kc) * 32 + lane) * 16;
#pragma unroll
  for (int i = 0; i < 16; ++i) {
    int kl = ((i < 8) ? i : i + 8) + hi * 8;
    o[i] = (bf16_t)xr[kl];
  }
}

// B-fragment (32x16 bf16): lane n(0..15)/hi, elem i -> k = ((i<8)?i:i+8)+hi*8
__global__ void k_wswz(const float* __restrict__ W, bf16_t* __restrict__ ws,
                       int Kch, int Nt, int Ncols) {
  int t = blockIdx.x * blockDim.x + threadIdx.x;
  if (t >= Kch * Nt * 32) return;
  int lane = t & 31, nt = (t >> 5) % Nt, kc = (t >> 5) / Nt;
  int n = lane & 15, hi = lane >> 4;
  bf16_t* o = ws + ((size_t)(kc * Nt + nt) * 32 + lane) * 16;
#pragma unroll
  for (int i = 0; i < 16; ++i)
    o[i] = (bf16_t)W[(size_t)(kc * 32 + ((i < 8) ? i : i + 8) + hi * 8) * Ncols + nt * 16 + n];
}

// ---------------- WMMA GEMMs ----------------
// h1[50000,256] = x @ W1 ; block = one 16-row tile, 8 waves x 2 N-tiles
__global__ void __launch_bounds__(256) k_gemm1(const bf16_t* __restrict__ xs,
                                               const bf16_t* __restrict__ w1s,
                                               float* __restrict__ h1) {
  int mt = blockIdx.x;
  int L = threadIdx.x & 31, wv = threadIdx.x >> 5;
  v8f a0 = {}, a1 = {};
#pragma unroll
  for (int kc = 0; kc < 4; ++kc) {
    v16bf av  = *(const v16bf*)(xs  + ((size_t)(mt * 4 + kc) * 32 + L) * 16);
    v16bf b0  = *(const v16bf*)(w1s + ((size_t)(kc * 16 + 2 * wv)     * 32 + L) * 16);
    v16bf b1v = *(const v16bf*)(w1s + ((size_t)(kc * 16 + 2 * wv + 1) * 32 + L) * 16);
    a0 = wmma_bf16(av, b0,  a0);
    a1 = wmma_bf16(av, b1v, a1);
  }
  int hi = L >> 4, n = L & 15;
  float* base = h1 + (size_t)(mt * 16 + hi * 8) * HIDALL + 2 * wv * 16 + n;
#pragma unroll
  for (int r = 0; r < 8; ++r) {
    base[(size_t)r * HIDALL]      = a0[r];
    base[(size_t)r * HIDALL + 16] = a1[r];
  }
}

// h2[50000,16] = act @ W2 ; one wave per 16-row tile, 8 K-chunks
__global__ void __launch_bounds__(256) k_gemm2(const bf16_t* __restrict__ as,
                                               const bf16_t* __restrict__ w2s,
                                               float* __restrict__ h2) {
  int gmt = blockIdx.x * 8 + (threadIdx.x >> 5);
  if (gmt >= MTILES) return;                    // wave-uniform: EXEC stays full
  int L = threadIdx.x & 31;
  v8f acc = {};
#pragma unroll
  for (int kc = 0; kc < 8; ++kc) {
    v16bf av = *(const v16bf*)(as  + ((size_t)(gmt * 8 + kc) * 32 + L) * 16);
    v16bf bv = *(const v16bf*)(w2s + ((size_t)kc * 32 + L) * 16);
    acc = wmma_bf16(av, bv, acc);
  }
  int hi = L >> 4, n = L & 15;
  float* base = h2 + (size_t)(gmt * 16 + hi * 8) * OUT_CH + n;
#pragma unroll
  for (int r = 0; r < 8; ++r) base[(size_t)r * OUT_CH] = acc[r];
}

// ---------------- attention logits ----------------
__global__ void k_al1(const float* __restrict__ h1, const float* __restrict__ asrc,
                      const float* __restrict__ adst, float* __restrict__ als,
                      float* __restrict__ ald) {
  int t = blockIdx.x * blockDim.x + threadIdx.x;
  if (t >= N_NODES * HEADS) return;
  int n = t >> 3, hd = t & 7;
  const float* hr = h1 + (size_t)n * HIDALL + hd * HID;
  float ss = 0.f, dd = 0.f;
#pragma unroll
  for (int c = 0; c < HID; ++c) {
    float v = hr[c];
    ss += v * asrc[hd * HID + c];
    dd += v * adst[hd * HID + c];
  }
  als[t] = ss; ald[t] = dd;
}

__global__ void k_al2(const float* __restrict__ h2, const float* __restrict__ asrc,
                      const float* __restrict__ adst, float* __restrict__ als,
                      float* __restrict__ ald) {
  int n = blockIdx.x * blockDim.x + threadIdx.x;
  if (n >= N_NODES) return;
  const float* hr = h2 + (size_t)n * OUT_CH;
  float ss = 0.f, dd = 0.f;
#pragma unroll
  for (int c = 0; c < OUT_CH; ++c) { ss += hr[c] * asrc[c]; dd += hr[c] * adst[c]; }
  als[n] = ss; ald[n] = dd;
}

// ---------------- layer-1 edge passes (H=8) ----------------
__global__ void k_e1a(const int* __restrict__ ei, const float* __restrict__ als,
                      const float* __restrict__ ald, float* __restrict__ e1,
                      unsigned* __restrict__ m1) {
  int t = blockIdx.x * blockDim.x + threadIdx.x;
  if (t >= E_TOT * HEADS) return;
  int e = t >> 3, hd = t & 7, s, d;
  edge_nodes(ei, e, s, d);
  float v = als[s * 8 + hd] + ald[d * 8 + hd];
  v = v > 0.f ? v : NEG_SLOPE * v;
  e1[t] = v;
  atomicMax(&m1[d * 8 + hd], f2ord(v));
}

__global__ void k_e1b(const int* __restrict__ ei, float* __restrict__ e1,
                      const unsigned* __restrict__ m1, float* __restrict__ s1) {
  int t = blockIdx.x * blockDim.x + threadIdx.x;
  if (t >= E_TOT * HEADS) return;
  int e = t >> 3, hd = t & 7, s, d;
  edge_nodes(ei, e, s, d);
  float ex = __expf(e1[t] - ord2f(m1[d * 8 + hd]));
  e1[t] = ex;
  atomicAdd(&s1[d * 8 + hd], ex);
}

// heavy pass: one wave32 per edge; lane = channel within head, 8 head iters
__global__ void __launch_bounds__(256) k_e1c(const int* __restrict__ ei,
                                             const float* __restrict__ h1,
                                             const float* __restrict__ e1,
                                             const float* __restrict__ s1,
                                             float* __restrict__ out1) {
  int wid = (blockIdx.x * blockDim.x + threadIdx.x) >> 5;
  if (wid >= E_TOT) return;
  int L = threadIdx.x & 31, s, d;
  edge_nodes(ei, wid, s, d);
  float aval = 0.f;
  if (L < 8) aval = e1[(size_t)wid * 8 + L] / (s1[d * 8 + L] + 1e-16f);
  const float* hs = h1 + (size_t)s * HIDALL;
  float* od = out1 + (size_t)d * HIDALL;
#pragma unroll
  for (int i = 0; i < 8; ++i) {
    float alpha = __shfl(aval, i, 32);
    atomicAdd(&od[i * 32 + L], hs[i * 32 + L] * alpha);
  }
}

// bias + ELU + bf16 A-fragment swizzle for layer 2
__global__ void k_act(const float* __restrict__ out1, const float* __restrict__ b1,
                      bf16_t* __restrict__ acts) {
  int t = blockIdx.x * blockDim.x + threadIdx.x;
  const int total = MTILES * 8 * 32;                       // 800000
  if (t >= total) return;
  int lane = t & 31, kc = (t >> 5) & 7, mt = t >> 8;
  int l = lane & 15, hi = lane >> 4;
  const float* orow = out1 + (size_t)(mt * 16 + l) * HIDALL + kc * 32;
  const float* brow = b1 + kc * 32;
  bf16_t* o = acts + ((size_t)(mt * 8 + kc) * 32 + lane) * 16;
#pragma unroll
  for (int i = 0; i < 16; ++i) {
    int kl = ((i < 8) ? i : i + 8) + hi * 8;
    float v = orow[kl] + brow[kl];
    v = v > 0.f ? v : (__expf(v) - 1.f);                   // ELU
    o[i] = (bf16_t)v;
  }
}

// ---------------- layer-2 edge passes (H=1) ----------------
__global__ void k_e2a(const int* __restrict__ ei, const float* __restrict__ als,
                      const float* __restrict__ ald, float* __restrict__ e2,
                      unsigned* __restrict__ m2) {
  int e = blockIdx.x * blockDim.x + threadIdx.x;
  if (e >= E_TOT) return;
  int s, d; edge_nodes(ei, e, s, d);
  float v = als[s] + ald[d];
  v = v > 0.f ? v : NEG_SLOPE * v;
  e2[e] = v;
  atomicMax(&m2[d], f2ord(v));
}

__global__ void k_e2b(const int* __restrict__ ei, float* __restrict__ e2,
                      const unsigned* __restrict__ m2, float* __restrict__ s2) {
  int e = blockIdx.x * blockDim.x + threadIdx.x;
  if (e >= E_TOT) return;
  int s, d; edge_nodes(ei, e, s, d);
  float ex = __expf(e2[e] - ord2f(m2[d]));
  e2[e] = ex;
  atomicAdd(&s2[d], ex);
}

__global__ void k_e2c(const int* __restrict__ ei, const float* __restrict__ h2,
                      const float* __restrict__ e2, const float* __restrict__ s2,
                      float* __restrict__ out2) {
  int t = blockIdx.x * blockDim.x + threadIdx.x;
  int e = t >> 4;
  if (e >= E_TOT) return;
  int c = t & 15, s, d;
  edge_nodes(ei, e, s, d);
  float alpha = e2[e] / (s2[d] + 1e-16f);
  atomicAdd(&out2[(size_t)d * OUT_CH + c], h2[(size_t)s * OUT_CH + c] * alpha);
}

// bias + log_softmax over 16 channels
__global__ void k_final(const float* __restrict__ out2, const float* __restrict__ b2,
                        float* __restrict__ out) {
  int n = blockIdx.x * blockDim.x + threadIdx.x;
  if (n >= N_NODES) return;
  float v[OUT_CH], mx = -1e30f;
#pragma unroll
  for (int c = 0; c < OUT_CH; ++c) {
    v[c] = out2[(size_t)n * OUT_CH + c] + b2[c];
    mx = fmaxf(mx, v[c]);
  }
  float sum = 0.f;
#pragma unroll
  for (int c = 0; c < OUT_CH; ++c) sum += __expf(v[c] - mx);
  float lse = mx + __logf(sum);
#pragma unroll
  for (int c = 0; c < OUT_CH; ++c) out[(size_t)n * OUT_CH + c] = v[c] - lse;
}

// ---------------- host-side orchestration ----------------
extern "C" void kernel_launch(void* const* d_in, const int* in_sizes, int n_in,
                              void* d_out, int out_size, void* d_ws, size_t ws_size,
                              hipStream_t stream) {
  const float* x   = (const float*)d_in[0];
  const int*   ei  = (const int*)  d_in[1];
  const float* W1  = (const float*)d_in[2];
  const float* as1 = (const float*)d_in[3];
  const float* ad1 = (const float*)d_in[4];
  const float* b1  = (const float*)d_in[5];
  const float* W2  = (const float*)d_in[6];
  const float* as2 = (const float*)d_in[7];
  const float* ad2 = (const float*)d_in[8];
  const float* b2  = (const float*)d_in[9];
  float* out = (float*)d_out;
  char*  ws  = (char*)d_ws;

  size_t off = 0;
  auto alloc = [&](size_t bytes) {
    size_t o = off;
    off += (bytes + 255) & ~(size_t)255;
    return o;
  };
  // zero-initialized group (single memset each launch)
  size_t o_m1   = alloc((size_t)N_NODES * 8 * 4);
  size_t o_s1   = alloc((size_t)N_NODES * 8 * 4);
  size_t o_out1 = alloc((size_t)N_NODES * HIDALL * 4);
  size_t o_m2   = alloc((size_t)N_NODES * 4);
  size_t o_s2   = alloc((size_t)N_NODES * 4);
  size_t o_out2 = alloc((size_t)N_NODES * OUT_CH * 4);
  size_t zero_bytes = off;
  // scratch (written before read each launch)
  size_t o_xs   = alloc((size_t)N_NODES * IN_CH * 2);
  size_t o_w1s  = alloc((size_t)IN_CH * HIDALL * 2);
  size_t o_w2s  = alloc((size_t)HIDALL * OUT_CH * 2);
  size_t o_h1   = alloc((size_t)N_NODES * HIDALL * 4);
  size_t o_als1 = alloc((size_t)N_NODES * 8 * 4);
  size_t o_ald1 = alloc((size_t)N_NODES * 8 * 4);
  size_t o_e1   = alloc((size_t)E_TOT * 8 * 4);
  size_t o_act  = alloc((size_t)N_NODES * HIDALL * 2);
  size_t o_h2   = alloc((size_t)N_NODES * OUT_CH * 4);
  size_t o_als2 = alloc((size_t)N_NODES * 4);
  size_t o_ald2 = alloc((size_t)N_NODES * 4);
  size_t o_e2   = alloc((size_t)E_TOT * 4);
  (void)ws_size; (void)in_sizes; (void)n_in; (void)out_size;

  unsigned* m1 = (unsigned*)(ws + o_m1);
  float* s1    = (float*)(ws + o_s1);
  float* out1  = (float*)(ws + o_out1);
  unsigned* m2 = (unsigned*)(ws + o_m2);
  float* s2    = (float*)(ws + o_s2);
  float* out2  = (float*)(ws + o_out2);
  bf16_t* xs   = (bf16_t*)(ws + o_xs);
  bf16_t* w1s  = (bf16_t*)(ws + o_w1s);
  bf16_t* w2s  = (bf16_t*)(ws + o_w2s);
  float* h1    = (float*)(ws + o_h1);
  float* als1  = (float*)(ws + o_als1);
  float* ald1  = (float*)(ws + o_ald1);
  float* e1    = (float*)(ws + o_e1);
  bf16_t* acts = (bf16_t*)(ws + o_act);
  float* h2    = (float*)(ws + o_h2);
  float* als2  = (float*)(ws + o_als2);
  float* ald2  = (float*)(ws + o_ald2);
  float* e2    = (float*)(ws + o_e2);

  hipMemsetAsync(ws, 0, zero_bytes, stream);   // graph-capture-safe memset node

  // operand prep
  k_xswz<<<(MTILES * 4 * 32 + 255) / 256, 256, 0, stream>>>(x, xs);
  k_wswz<<<(4 * 16 * 32 + 255) / 256, 256, 0, stream>>>(W1, w1s, 4, 16, HIDALL);
  k_wswz<<<1, 256, 0, stream>>>(W2, w2s, 8, 1, OUT_CH);

  // layer 1
  k_gemm1<<<MTILES, 256, 0, stream>>>(xs, w1s, h1);
  k_al1<<<(N_NODES * HEADS + 255) / 256, 256, 0, stream>>>(h1, as1, ad1, als1, ald1);
  k_e1a<<<(E_TOT * HEADS + 255) / 256, 256, 0, stream>>>(ei, als1, ald1, e1, m1);
  k_e1b<<<(E_TOT * HEADS + 255) / 256, 256, 0, stream>>>(ei, e1, m1, s1);
  k_e1c<<<E_TOT / 8, 256, 0, stream>>>(ei, h1, e1, s1, out1);
  k_act<<<(MTILES * 8 * 32 + 255) / 256, 256, 0, stream>>>(out1, b1, acts);

  // layer 2
  k_gemm2<<<(MTILES + 7) / 8, 256, 0, stream>>>(acts, w2s, h2);
  k_al2<<<(N_NODES + 255) / 256, 256, 0, stream>>>(h2, as2, ad2, als2, ald2);
  k_e2a<<<(E_TOT + 255) / 256, 256, 0, stream>>>(ei, als2, ald2, e2, m2);
  k_e2b<<<(E_TOT + 255) / 256, 256, 0, stream>>>(ei, e2, m2, s2);
  k_e2c<<<(E_TOT * 16) / 256, 256, 0, stream>>>(ei, h2, e2, s2, out2);
  k_final<<<(N_NODES + 255) / 256, 256, 0, stream>>>(out2, b2, out);
}
